// MHA_65326452572677
// MI455X (gfx1250) — matrix operand order repaired and checked by reading, not compiled, not run on hardware
//
#include <hip/hip_runtime.h>

// Fused causal MHA forward for MI455X (gfx1250, wave32):
//   - v_wmma_f32_16x16x32_bf16 for all matmuls (f32 accumulate)
//   - GLOBAL_LOAD_ASYNC_TO_LDS_B128 + s_wait_asynccnt (attention K staging)
//   - TENSOR_LOAD_TO_LDS (TDM) + s_wait_tensorcnt (output-proj A staging)
// B=2, T=2048, D=2048, H=16, Dh=128.

#define BB 2
#define TT 2048
#define DD 2048
#define HH 16
#define DH 128

typedef __attribute__((ext_vector_type(16))) __bf16 v16bf;
typedef __attribute__((ext_vector_type(8)))  float  v8f;
typedef unsigned int u32x4 __attribute__((ext_vector_type(4)));
typedef int          i32x4 __attribute__((ext_vector_type(4)));
typedef int          i32x8 __attribute__((ext_vector_type(8)));

union FragBF { v16bf v; unsigned int u[8]; };
union AccF   { v8f   v; float f[8]; };

#if defined(__has_builtin)
#  if __has_builtin(__builtin_amdgcn_tensor_load_to_lds) && \
      __has_builtin(__builtin_amdgcn_s_wait_tensorcnt)
#    define HAVE_TDM 1
#  endif
#endif
#ifndef HAVE_TDM
#  define HAVE_TDM 0
#endif

// fp32 -> bf16 round-to-nearest-even
__device__ __forceinline__ unsigned short f2bf(float x) {
    unsigned int u = __float_as_uint(x);
    u += 0x7FFFu + ((u >> 16) & 1u);
    return (unsigned short)(u >> 16);
}
__device__ __forceinline__ unsigned int pack2bf(float lo, float hi) {
    return (unsigned int)f2bf(lo) | ((unsigned int)f2bf(hi) << 16);
}

// CDNA5 async copy: 16B global -> LDS, tracked by ASYNCcnt (no VGPR data path).
__device__ __forceinline__ void async_b128(void* lds, const void* gsrc) {
    unsigned l = (unsigned)(size_t)lds;     // low 32 bits of generic ptr = LDS offset
    asm volatile("global_load_async_to_lds_b128 %0, %1, off"
                 :: "v"(l), "v"(gsrc) : "memory");
}
__device__ __forceinline__ void wait_async_le2() { asm volatile("s_wait_asynccnt 2" ::: "memory"); }
__device__ __forceinline__ void wait_async_0()   { asm volatile("s_wait_asynccnt 0" ::: "memory"); }

#if HAVE_TDM
// TDM 2-D bf16 tile load: tileW x tileH elements from a strideElems-pitch tensor,
// with LDS row padding (pad codes per D# group1 encoding). Issued once per wave;
// EXEC ignored by tensor ops, so a single active lane is enough.
// This toolchain's builtin takes 6 args: (g0 u32x4, g1 i32x8, g2 i32x4, g3 i32x4,
// extra i32x8, cpol i32); groups 2/3 unused for <=2-D tensors -> zero-filled.
__device__ __forceinline__ void tdm_load_tile_bf16(unsigned ldsAddr, const void* gsrc,
                                                   unsigned tileW, unsigned tileH,
                                                   unsigned strideElems,
                                                   unsigned padIntervalCode,
                                                   unsigned padAmountCode)
{
    const unsigned long long ga = (unsigned long long)(size_t)gsrc;
    u32x4 g0;
    g0[0] = 1u;                                                 // count=1, user D#
    g0[1] = ldsAddr;                                            // lds_addr
    g0[2] = (unsigned)ga;                                       // global_addr[31:0]
    g0[3] = (unsigned)((ga >> 32) & 0x01FFFFFFu) | (2u << 30);  // addr[56:32] | type=2
    i32x8 g1;
    g1[0] = (int)((1u << 16)                                    // data_size = 2B
                | (1u << 20)                                    // pad_enable
                | (padIntervalCode << 22) | (padAmountCode << 25));
    g1[1] = (int)(tileW << 16);                                 // tensor_dim0 = tileW
    g1[2] = (int)(tileH << 16);                                 // tensor_dim1 = tileH
    g1[3] = (int)(tileW << 16);                                 // tile_dim0  = tileW
    g1[4] = (int)tileH;                                         // tile_dim1  = tileH
    g1[5] = (int)strideElems;                                   // tensor_dim0_stride
    g1[6] = 0;
    g1[7] = 0;
    const i32x4 z4 = {0, 0, 0, 0};
    const i32x8 z8 = {0, 0, 0, 0, 0, 0, 0, 0};
    __builtin_amdgcn_tensor_load_to_lds(g0, g1, z4, z4, z8, 0);
}
#endif

// ---------------------------------------------------------------------------
// C = A (MxK) * B^T, B stored [N][K] (torch Linear weight layout).
// Block tile 128x128, K-step 32, 8 waves = 4(rows*32) x 2(cols*64); each wave
// a 2x4 grid of 16x16 WMMA accumulators. Software-pipelined staging:
//   MODE 0: A = fp32 x   (reg-pipelined, cvt to bf16) -> bf16 out [B,H,T,Dh], scaled
//   MODE 1: A = bf16 ctx (TDM tensor_load_to_lds, double-buffered) -> fp32 [M][N]
// W is always fp32, reg-pipelined with on-the-fly bf16 conversion.
// ---------------------------------------------------------------------------
template<int MODE>
__global__ __launch_bounds__(256)
void gemm_abt(const float* __restrict__ A32,
              const unsigned short* __restrict__ A16,
              const float* __restrict__ W,
              unsigned short* __restrict__ outQKV,
              float* __restrict__ out32,
              float oscale)
{
    constexpr int NXB = (MODE == 1) ? 2 : 1;
    __shared__ alignas(16) unsigned int Xs[NXB][128][20];  // 32 bf16/row in words 0..15
    __shared__ alignas(16) unsigned int Ws[128][20];

    const int tid  = threadIdx.x;
    const int lane = tid & 31, wave = tid >> 5;
    const int lr   = lane & 15, hs = lane >> 4;
    const int wm   = wave & 3,  wn = wave >> 2;
    const int rowBase = blockIdx.x * 128;
    const int colBase = blockIdx.y * 128;

    const int srow = tid >> 3, sg = tid & 7;   // fp32 staging: 4 rows of float4 each
    const int arow = tid >> 2, ag = tid & 3;   // bf16 async staging: 2 rows of 16B each
    (void)arow; (void)ag;

    // ---- prologue: tile 0 in flight ----
    float4 wreg[4], xreg[4];
    #pragma unroll
    for (int i = 0; i < 4; ++i)
        wreg[i] = *(const float4*)(W + (size_t)(colBase + srow + 32 * i) * DD + sg * 4);
    if (MODE == 0) {
        #pragma unroll
        for (int i = 0; i < 4; ++i)
            xreg[i] = *(const float4*)(A32 + (size_t)(rowBase + srow + 32 * i) * DD + sg * 4);
    } else {
#if HAVE_TDM
        if (tid == 0)
            tdm_load_tile_bf16((unsigned)(size_t)&Xs[0][0][0],
                               A16 + (size_t)rowBase * DD,
                               /*tileW=*/32, /*tileH=*/128, /*stride=*/DD,
                               /*padInterval: 16 DW=*/3, /*padAmount: 4 DW=*/3);
#else
        async_b128(&Xs[0][arow][ag * 4],      A16 + (size_t)(rowBase + arow) * DD + ag * 8);
        async_b128(&Xs[0][arow + 64][ag * 4], A16 + (size_t)(rowBase + arow + 64) * DD + ag * 8);
#endif
    }

    AccF acc[2][4];
    #pragma unroll
    for (int i = 0; i < 2; ++i)
        #pragma unroll
        for (int j = 0; j < 4; ++j)
            #pragma unroll
            for (int r = 0; r < 8; ++r) acc[i][j].f[r] = 0.0f;

    int cb = 0;
    for (int k0 = 0; k0 < DD; k0 += 32) {
        const bool hasNext = (k0 + 32) < DD;

        // ---- commit current W (and X for MODE 0) registers to LDS ----
        #pragma unroll
        for (int i = 0; i < 4; ++i) {
            Ws[srow + 32 * i][sg * 2 + 0] = pack2bf(wreg[i].x, wreg[i].y);
            Ws[srow + 32 * i][sg * 2 + 1] = pack2bf(wreg[i].z, wreg[i].w);
        }
        if (MODE == 0) {
            #pragma unroll
            for (int i = 0; i < 4; ++i) {
                Xs[0][srow + 32 * i][sg * 2 + 0] = pack2bf(xreg[i].x, xreg[i].y);
                Xs[0][srow + 32 * i][sg * 2 + 1] = pack2bf(xreg[i].z, xreg[i].w);
            }
        }
        // ---- issue next tile (overlaps with this tile's WMMAs) ----
        if (hasNext) {
            const int kn = k0 + 32;
            #pragma unroll
            for (int i = 0; i < 4; ++i)
                wreg[i] = *(const float4*)(W + (size_t)(colBase + srow + 32 * i) * DD + kn + sg * 4);
            if (MODE == 0) {
                #pragma unroll
                for (int i = 0; i < 4; ++i)
                    xreg[i] = *(const float4*)(A32 + (size_t)(rowBase + srow + 32 * i) * DD + kn + sg * 4);
            } else {
#if HAVE_TDM
                if (tid == 0)
                    tdm_load_tile_bf16((unsigned)(size_t)&Xs[cb ^ 1][0][0],
                                       A16 + (size_t)rowBase * DD + kn,
                                       32, 128, DD, 3, 3);
#else
                const int nb = cb ^ 1;
                async_b128(&Xs[nb][arow][ag * 4],      A16 + (size_t)(rowBase + arow) * DD + kn + ag * 8);
                async_b128(&Xs[nb][arow + 64][ag * 4], A16 + (size_t)(rowBase + arow + 64) * DD + kn + ag * 8);
#endif
            }
        }
        if (MODE == 1) {
#if HAVE_TDM
            if (wave == 0) {
                if (hasNext) __builtin_amdgcn_s_wait_tensorcnt((unsigned short)1);
                else         __builtin_amdgcn_s_wait_tensorcnt((unsigned short)0);
            }
#else
            if (hasNext) wait_async_le2(); else wait_async_0();
#endif
        }
        __syncthreads();

        // ---- fragments (contiguous, 16B-aligned -> ds_load_b128) + 8 WMMA ----
        const int xb = (MODE == 1) ? cb : 0;
        FragBF a[2], b[4];
        #pragma unroll
        for (int i = 0; i < 2; ++i) {
            const unsigned int* xr = Xs[xb][wm * 32 + i * 16 + lr];
            #pragma unroll
            for (int q = 0; q < 4; ++q) {
                a[i].u[q]     = xr[hs * 4 + q];
                a[i].u[4 + q] = xr[8 + hs * 4 + q];
            }
        }
        #pragma unroll
        for (int j = 0; j < 4; ++j) {
            const unsigned int* wr = Ws[wn * 64 + j * 16 + lr];
            #pragma unroll
            for (int q = 0; q < 4; ++q) { b[j].u[q] = wr[hs * 4 + q]; b[j].u[4 + q] = wr[8 + hs * 4 + q]; }
        }
        #pragma unroll
        for (int i = 0; i < 2; ++i)
            #pragma unroll
            for (int j = 0; j < 4; ++j)
                acc[i][j].v = __builtin_amdgcn_wmma_f32_16x16x32_bf16(
                    false, a[i].v, false, b[j].v, (short)0, acc[i][j].v, false, false);
        __syncthreads();
        cb ^= 1;
    }

    // ---- epilogue ----
    #pragma unroll
    for (int i = 0; i < 2; ++i) {
        #pragma unroll
        for (int j = 0; j < 4; ++j) {
            const int n = colBase + wn * 64 + j * 16 + lr;
            #pragma unroll
            for (int r = 0; r < 8; ++r) {
                const int m = rowBase + wm * 32 + i * 16 + r + 8 * hs;
                const float val = acc[i][j].f[r];
                if (MODE == 0) {
                    const int bb = m / TT, t = m % TT;
                    const int hh = n / DH, d = n % DH;
                    outQKV[((size_t)(bb * HH + hh) * TT + t) * DH + d] = f2bf(val * oscale);
                } else {
                    out32[(size_t)m * DD + n] = val;
                }
            }
        }
    }
}

// ---------------------------------------------------------------------------
// Flash attention: block = 128 q rows of one (b,h); wave w owns rows [w*16,+16).
// K tiles (32x128 bf16) async-copied global->LDS (ASYNCcnt), double-buffered;
// V staged transposed via registers (reg-pipelined); online softmax (scale
// pre-folded into Q); causal mask analytic (== reference additive mask).
// ctx written bf16 to [B,T,D].
// ---------------------------------------------------------------------------
__global__ __launch_bounds__(256)
void flash_attn(const unsigned short* __restrict__ Qb,
                const unsigned short* __restrict__ Kb,
                const unsigned short* __restrict__ Vb,
                unsigned short* __restrict__ Ctx)
{
    __shared__ alignas(16) unsigned int   Ks[2][32][68];  // 128 bf16/row in words 0..63
    __shared__ alignas(16) unsigned short Vs[128][40];    // transposed: [dh][k]
    __shared__ alignas(16) unsigned short Ps[8][16][40];  // per-wave P tile 16x32

    const int tid  = threadIdx.x;
    const int lane = tid & 31, wave = tid >> 5;
    const int lr   = lane & 15, hs = lane >> 4;
    const int bh   = blockIdx.y;
    const int qBlock = blockIdx.x * 128;
    const int qRow0  = qBlock + wave * 16;

    const unsigned short* Qh = Qb + (size_t)bh * TT * DH;
    const unsigned short* Kh = Kb + (size_t)bh * TT * DH;
    const unsigned short* Vh = Vb + (size_t)bh * TT * DH;

    const int krow = tid >> 4, kg = tid & 15;  // K async: 2 rows of 16B per thread
    const int vrow = tid >> 6, vc = tid & 63;  // V: 8 bf16-pairs per thread

    // Q fragments (scale already folded in at projection)
    FragBF qf[4];
    {
        const unsigned int* qr = (const unsigned int*)(Qh + (size_t)(qRow0 + lr) * DH);
        #pragma unroll
        for (int kk = 0; kk < 4; ++kk)
            #pragma unroll
            for (int q = 0; q < 4; ++q) {
                qf[kk].u[q]     = qr[kk * 16 + hs * 4 + q];
                qf[kk].u[4 + q] = qr[kk * 16 + 8 + hs * 4 + q];
            }
    }

    AccF o[8];
    #pragma unroll
    for (int j = 0; j < 8; ++j)
        #pragma unroll
        for (int r = 0; r < 8; ++r) o[j].f[r] = 0.0f;
    float m8[8], l8[8];
    #pragma unroll
    for (int r = 0; r < 8; ++r) { m8[r] = -1e30f; l8[r] = 0.0f; }

    const int nkt = (qBlock + 128) / 32;   // causal k-tile limit (uniform per block)

    // ---- prologue: tile 0 in flight ----
    async_b128(&Ks[0][krow][kg * 4],      Kh + (size_t)krow * DH + kg * 8);
    async_b128(&Ks[0][krow + 16][kg * 4], Kh + (size_t)(krow + 16) * DH + kg * 8);
    unsigned int vreg[8];
    #pragma unroll
    for (int i = 0; i < 8; ++i)
        vreg[i] = *(const unsigned int*)(Vh + (size_t)(vrow + 4 * i) * DH + vc * 2);

    int cb = 0;
    for (int kt = 0; kt < nkt; ++kt) {
        const int kBase = kt * 32;
        const bool hasNext = (kt + 1) < nkt;

        // ---- commit V tile (transposed) to LDS ----
        #pragma unroll
        for (int i = 0; i < 8; ++i) {
            Vs[vc * 2 + 0][vrow + 4 * i] = (unsigned short)(vreg[i] & 0xFFFFu);
            Vs[vc * 2 + 1][vrow + 4 * i] = (unsigned short)(vreg[i] >> 16);
        }
        // ---- issue next tile ----
        if (hasNext) {
            const int kn = kBase + 32;
            const int nb = cb ^ 1;
            async_b128(&Ks[nb][krow][kg * 4],      Kh + (size_t)(kn + krow) * DH + kg * 8);
            async_b128(&Ks[nb][krow + 16][kg * 4], Kh + (size_t)(kn + krow + 16) * DH + kg * 8);
            #pragma unroll
            for (int i = 0; i < 8; ++i)
                vreg[i] = *(const unsigned int*)(Vh + (size_t)(kn + vrow + 4 * i) * DH + vc * 2);
        }
        if (hasNext) wait_async_le2(); else wait_async_0();
        __syncthreads();

        // ---- S = Q K^T for both 16-col subtiles ----
        AccF s0, s1;
        #pragma unroll
        for (int r = 0; r < 8; ++r) { s0.f[r] = 0.0f; s1.f[r] = 0.0f; }
        const unsigned int* kr0 = Ks[cb][lr];
        const unsigned int* kr1 = Ks[cb][16 + lr];
        #pragma unroll
        for (int kk = 0; kk < 4; ++kk) {
            FragBF b0, b1;
            #pragma unroll
            for (int q = 0; q < 4; ++q) { b0.u[q] = kr0[kk * 16 + hs * 4 + q]; b0.u[4 + q] = kr0[kk * 16 + 8 + hs * 4 + q];
                                          b1.u[q] = kr1[kk * 16 + hs * 4 + q]; b1.u[4 + q] = kr1[kk * 16 + 8 + hs * 4 + q]; }
            s0.v = __builtin_amdgcn_wmma_f32_16x16x32_bf16(
                false, qf[kk].v, false, b0.v, (short)0, s0.v, false, false);
            s1.v = __builtin_amdgcn_wmma_f32_16x16x32_bf16(
                false, qf[kk].v, false, b1.v, (short)0, s1.v, false, false);
        }

        // ---- merged online softmax over 32 k-columns ----
        #pragma unroll
        for (int r = 0; r < 8; ++r) {
            const int qrow = qRow0 + r + 8 * hs;
            float a0 = s0.f[r]; if (kBase + lr > qrow)      a0 = -1e30f;
            float a1 = s1.f[r]; if (kBase + 16 + lr > qrow) a1 = -1e30f;
            float mx = fmaxf(a0, a1);
            mx = fmaxf(mx, __shfl_xor(mx, 1));
            mx = fmaxf(mx, __shfl_xor(mx, 2));
            mx = fmaxf(mx, __shfl_xor(mx, 4));
            mx = fmaxf(mx, __shfl_xor(mx, 8));
            const float mN    = fmaxf(m8[r], mx);
            const float alpha = __expf(m8[r] - mN);
            const float p0    = __expf(a0 - mN);
            const float p1    = __expf(a1 - mN);
            float ps = p0 + p1;
            ps += __shfl_xor(ps, 1);
            ps += __shfl_xor(ps, 2);
            ps += __shfl_xor(ps, 4);
            ps += __shfl_xor(ps, 8);
            l8[r] = l8[r] * alpha + ps;
            m8[r] = mN;
            #pragma unroll
            for (int j = 0; j < 8; ++j) o[j].f[r] *= alpha;
            Ps[wave][r + 8 * hs][lr]      = f2bf(p0);
            Ps[wave][r + 8 * hs][16 + lr] = f2bf(p1);
        }
        // No barrier needed: Ps bounce is wave-private and DS ops from one wave
        // are executed in order (DScnt in-order).

        // ---- O += P V ----
        FragBF pa;
        {
            const unsigned int* pr = (const unsigned int*)&Ps[wave][lr][0];
            #pragma unroll
            for (int q = 0; q < 4; ++q) {
                pa.u[q]     = pr[hs * 4 + q];
                pa.u[4 + q] = pr[8 + hs * 4 + q];
            }
        }
        #pragma unroll
        for (int j = 0; j < 8; ++j) {
            FragBF bv;
            const unsigned int* vru = (const unsigned int*)&Vs[j * 16 + lr][0];
            #pragma unroll
            for (int q = 0; q < 4; ++q) { bv.u[q] = vru[hs * 4 + q]; bv.u[4 + q] = vru[8 + hs * 4 + q]; }
            o[j].v = __builtin_amdgcn_wmma_f32_16x16x32_bf16(
                false, pa.v, false, bv.v, (short)0, o[j].v, false, false);
        }
        __syncthreads();   // protect Vs reuse next iteration
        cb ^= 1;
    }

    // ---- normalize and store ctx bf16 into [B, T, D] ----
    const int b = bh / HH, h = bh % HH;
    #pragma unroll
    for (int j = 0; j < 8; ++j) {
        const int dh = j * 16 + lr;
        #pragma unroll
        for (int r = 0; r < 8; ++r) {
            const int t = qRow0 + r + 8 * hs;
            const float val = o[j].f[r] / l8[r];
            Ctx[((size_t)(b * TT + t)) * DD + h * DH + dh] = f2bf(val);
        }
    }
}

// ---------------------------------------------------------------------------
extern "C" void kernel_launch(void* const* d_in, const int* in_sizes, int n_in,
                              void* d_out, int out_size, void* d_ws, size_t ws_size,
                              hipStream_t stream)
{
    const float* x  = (const float*)d_in[0];
    const float* Wq = (const float*)d_in[1];
    const float* Wk = (const float*)d_in[2];
    const float* Wv = (const float*)d_in[3];
    const float* Wo = (const float*)d_in[4];
    // d_in[5] = attn_mask: causal, applied analytically inside flash_attn.

    unsigned short* ws = (unsigned short*)d_ws;
    const size_t per = (size_t)BB * HH * TT * DH;   // 8,388,608 bf16 elems
    unsigned short* Qb = ws;
    unsigned short* Kb = ws + per;
    unsigned short* Vb = ws + 2 * per;
    unsigned short* Cx = ws + 3 * per;

    const float qscale = 0.08838834764831845f;      // 1/sqrt(Dh), folded into Q

    dim3 blk(256);
    dim3 g1(BB * TT / 128, DD / 128);    // 32 x 16

    gemm_abt<0><<<g1, blk, 0, stream>>>(x, nullptr, Wq, Qb, nullptr, qscale);
    gemm_abt<0><<<g1, blk, 0, stream>>>(x, nullptr, Wk, Kb, nullptr, 1.0f);
    gemm_abt<0><<<g1, blk, 0, stream>>>(x, nullptr, Wv, Vb, nullptr, 1.0f);

    dim3 g2(TT / 128, BB * HH);          // 16 x 32
    flash_attn<<<g2, blk, 0, stream>>>(Qb, Kb, Vb, Cx);

    gemm_abt<1><<<g1, blk, 0, stream>>>(nullptr, Cx, Wo, nullptr, (float*)d_out, 1.0f);
}
